// EnergyOutput_12189117186315
// MI455X (gfx1250) — compile-verified, matrix operand order checked by reading, not run on hardware
//
#include <hip/hip_runtime.h>
#include <hip/hip_bf16.h>

// ---------------------------------------------------------------------------
// EnergyOutput: fused 3-layer MLP (256->256->256->1, SiLU) + segment-sum pool.
// bf16 WMMA (v_wmma_f32_16x16x32_bf16), f32 accumulation, fully fused.
// Round 3: M=32 rows per wave (two A tiles in VGPRs) so every B fragment
// feeds TWO WMMAs -> halves B-fragment L0 traffic per MAC (the bottleneck),
// and gives two independent accumulator chains for free.
// ---------------------------------------------------------------------------

typedef __bf16 bf16_t;
typedef __attribute__((ext_vector_type(16))) __bf16 v16bf;
typedef __attribute__((ext_vector_type(8)))  float  v8f;

#define FEAT    256
#define WAVES   4        // waves per block
#define M_WAVE  32       // atom rows per wave (two 16-row WMMA tiles)
#define TILE_M  (WAVES * M_WAVE)   // 128 rows per block -> 64 KB LDS tile
#define SCALE_F 5.992277830325989f
#define SHIFT_F -406274.63784969115f

union FragB { v16bf v; uint4 q[2]; };

// Prep: W1T/W2T[n*256+k] = bf16(W[k*256+n])  (transposed so B-fragment loads
// are contiguous 16B chunks per lane), and d_out[j] = SHIFT.
__global__ void energy_prep_kernel(const float* __restrict__ W1,
                                   const float* __restrict__ W2,
                                   bf16_t* __restrict__ W1T,
                                   bf16_t* __restrict__ W2T,
                                   float* __restrict__ out, int nmol) {
    int i = blockIdx.x * blockDim.x + threadIdx.x;
    if (i < FEAT * FEAT) {
        int k = i >> 8;
        int n = i & 255;
        W1T[n * FEAT + k] = (bf16_t)W1[i];
        W2T[n * FEAT + k] = (bf16_t)W2[i];
    }
    if (i < nmol) out[i] = SHIFT_F;
}

// Fast SiLU: v_exp_f32 + v_add + v_rcp_f32 + v_mul.
__device__ __forceinline__ float silu_f(float x) {
    return x * __builtin_amdgcn_rcpf(1.0f + __expf(-x));
}

// Swizzled LDS address (in bf16 elements) for activation tile element (m, k):
// 16-byte chunks XOR'ed by row so 16 consecutive rows hit distinct banks.
__device__ __forceinline__ int lds_idx(int m, int k) {
    return m * FEAT + ((((k >> 3) ^ (m & 15)) << 3) | (k & 7));
}

__global__ __launch_bounds__(WAVES * 32) void energy_mlp_pool_kernel(
    const float*  __restrict__ X,      // [n_atoms, 256] fp32
    const int*    __restrict__ batch,  // [n_atoms] sorted molecule ids
    const bf16_t* __restrict__ W1T,    // [256,256] bf16, transposed
    const float*  __restrict__ b1,     // [256]
    const bf16_t* __restrict__ W2T,    // [256,256] bf16, transposed
    const float*  __restrict__ b2,     // [256]
    const float*  __restrict__ W3,     // [256]
    const float*  __restrict__ b3,     // [1]
    float*        __restrict__ out,    // [N_MOL], pre-initialized to SHIFT
    int n_atoms)
{
    __shared__ bf16_t Hs[TILE_M * FEAT];   // 64 KB: layer-1 activations (bf16)

    const int tid  = threadIdx.x;
    const int wave = tid >> 5;            // 0..3
    const int lane = tid & 31;
    const int lh   = lane & 15;           // lane within 16-half
    const int half = lane >> 4;           // 0/1
    const int koff = half << 3;           // K sub-offset per ISA A/B layout
    const int mloc0 = wave * M_WAVE;      // wave's local row base in tile
    const int m0    = blockIdx.x * TILE_M + mloc0;  // global row base

    // ------- Layer 1 A-fragments: 2 tiles x 16 rows straight from global ---
    // lane covers row (tile base + lh); K = koff+{0..7} and koff+{16..23}.
    v16bf a1[2][8];
    #pragma unroll
    for (int t = 0; t < 2; ++t) {
        int arow = m0 + t * 16 + lh;
        arow = (arow < n_atoms) ? arow : (n_atoms - 1);   // clamp (masked later)
        const float* xrow = X + (size_t)arow * FEAT;
        #pragma unroll
        for (int ks = 0; ks < 8; ++ks) {
            const float* p = xrow + ks * 32 + koff;
            #pragma unroll
            for (int j = 0; j < 8; ++j) {
                a1[t][ks][j]     = (bf16_t)p[j];
                a1[t][ks][j + 8] = (bf16_t)p[16 + j];
            }
        }
    }

    // ---------------- Layer 1: H = silu(X @ W1 + b1) -> LDS (bf16) ---------
    // One B fragment feeds two WMMAs (tile 0 / tile 1): 2x B reuse + 2 chains.
    #pragma unroll 1
    for (int nt = 0; nt < 16; ++nt) {
        const int n = (nt << 4) + lh;            // lane's output column
        const float bias = b1[n];
        v8f acc0 = { bias, bias, bias, bias, bias, bias, bias, bias };
        v8f acc1 = acc0;
        const bf16_t* wrow = W1T + (size_t)n * FEAT;
        #pragma unroll
        for (int ks = 0; ks < 8; ++ks) {
            FragB bu;
            const bf16_t* q = wrow + ks * 32 + koff;
            bu.q[0] = *(const uint4*)(q);
            bu.q[1] = *(const uint4*)(q + 16);
            acc0 = __builtin_amdgcn_wmma_f32_16x16x32_bf16(
                false, a1[0][ks], false, bu.v, (short)0, acc0, false, false);
            acc1 = __builtin_amdgcn_wmma_f32_16x16x32_bf16(
                false, a1[1][ks], false, bu.v, (short)0, acc1, false, false);
        }
        #pragma unroll
        for (int v = 0; v < 8; ++v) {
            const int mr = mloc0 + (half << 3) + v;   // C layout: M = 8*(lane/16)+v
            Hs[lds_idx(mr,      n)] = (bf16_t)silu_f(acc0[v]);
            Hs[lds_idx(mr + 16, n)] = (bf16_t)silu_f(acc1[v]);
        }
    }
    __syncthreads();

    // ---------------- Layer 2 A-fragments from LDS (ds_read_b128) ----------
    v16bf a2[2][8];
    #pragma unroll
    for (int t = 0; t < 2; ++t) {
        const int m = mloc0 + t * 16 + lh;
        #pragma unroll
        for (int ks = 0; ks < 8; ++ks) {
            const int c0 = (ks << 2) + half;          // (ks*32+koff)/8
            FragB au;
            au.q[0] = *(const uint4*)&Hs[m * FEAT + (((c0)     ^ (m & 15)) << 3)];
            au.q[1] = *(const uint4*)&Hs[m * FEAT + (((c0 + 2) ^ (m & 15)) << 3)];
            a2[t][ks] = au.v;
        }
    }

    // -------- Layer 2 + fused layer 3: psum[t][v] += silu(h2) * W3[n] ------
    float psum[2][8] = {{0.f,0.f,0.f,0.f,0.f,0.f,0.f,0.f},
                        {0.f,0.f,0.f,0.f,0.f,0.f,0.f,0.f}};
    #pragma unroll 1
    for (int nt = 0; nt < 16; ++nt) {
        const int n = (nt << 4) + lh;
        const float bias = b2[n];
        v8f acc0 = { bias, bias, bias, bias, bias, bias, bias, bias };
        v8f acc1 = acc0;
        const bf16_t* wrow = W2T + (size_t)n * FEAT;
        #pragma unroll
        for (int ks = 0; ks < 8; ++ks) {
            FragB bu;
            const bf16_t* q = wrow + ks * 32 + koff;
            bu.q[0] = *(const uint4*)(q);
            bu.q[1] = *(const uint4*)(q + 16);
            acc0 = __builtin_amdgcn_wmma_f32_16x16x32_bf16(
                false, a2[0][ks], false, bu.v, (short)0, acc0, false, false);
            acc1 = __builtin_amdgcn_wmma_f32_16x16x32_bf16(
                false, a2[1][ks], false, bu.v, (short)0, acc1, false, false);
        }
        const float w3n = W3[n];
        #pragma unroll
        for (int v = 0; v < 8; ++v) {
            psum[0][v] = __builtin_fmaf(silu_f(acc0[v]), w3n, psum[0][v]);
            psum[1][v] = __builtin_fmaf(silu_f(acc1[v]), w3n, psum[1][v]);
        }
    }

    // Reduce over the 16 lanes of each half (N=lane in C/D layout).
    #pragma unroll
    for (int t = 0; t < 2; ++t) {
        #pragma unroll
        for (int v = 0; v < 8; ++v) {
            float s = psum[t][v];
            s += __shfl_xor(s, 1, 32);
            s += __shfl_xor(s, 2, 32);
            s += __shfl_xor(s, 4, 32);
            s += __shfl_xor(s, 8, 32);
            psum[t][v] = s;
        }
    }

    // Lanes 0 and 16 own M = {0..7} and {8..15} of each tile respectively.
    if (lh == 0) {
        const float bb = b3[0];
        #pragma unroll
        for (int t = 0; t < 2; ++t) {
            #pragma unroll
            for (int v = 0; v < 8; ++v) {
                const int row = m0 + t * 16 + (half << 3) + v;
                if (row < n_atoms) {
                    const float e = (psum[t][v] + bb) * SCALE_F;
                    __hip_atomic_fetch_add(&out[batch[row]], e,
                                           __ATOMIC_RELAXED,
                                           __HIP_MEMORY_SCOPE_AGENT);
                }
            }
        }
    }
}

extern "C" void kernel_launch(void* const* d_in, const int* in_sizes, int n_in,
                              void* d_out, int out_size, void* d_ws, size_t ws_size,
                              hipStream_t stream) {
    const float* X     = (const float*)d_in[0];   // atom_node [N,256] f32
    const int*   batch = (const int*)  d_in[1];   // [N] int
    const float* W1    = (const float*)d_in[2];
    const float* b1    = (const float*)d_in[3];
    const float* W2    = (const float*)d_in[4];
    const float* b2    = (const float*)d_in[5];
    const float* W3    = (const float*)d_in[6];
    const float* b3    = (const float*)d_in[7];
    float* out = (float*)d_out;

    const int n_atoms = in_sizes[0] / FEAT;

    bf16_t* W1T = (bf16_t*)d_ws;                  // 128 KB
    bf16_t* W2T = W1T + FEAT * FEAT;              // 128 KB

    energy_prep_kernel<<<(FEAT * FEAT + 255) / 256, 256, 0, stream>>>(
        W1, W2, W1T, W2T, out, out_size);

    const int nblocks = (n_atoms + TILE_M - 1) / TILE_M;
    energy_mlp_pool_kernel<<<nblocks, WAVES * 32, 0, stream>>>(
        X, batch, W1T, b1, W2T, b2, W3, b3, out, n_atoms);
}
